// MultiHeadAttention_2319282340107
// MI455X (gfx1250) — matrix operand order repaired and checked by reading, not compiled, not run on hardware
//
#include <hip/hip_runtime.h>

// ---------------------------------------------------------------------------
// MHA forward for MI455X (gfx1250, wave32, WMMA bf16 16x16x32).
// Memory-bound by the mandatory 1.07GB attn_prob write; score strips are kept
// LDS-resident (16 rows x 4096 keys x fp32 = 256KB of the 320KB/WGP LDS) so
// probabilities touch HBM exactly once (via nontemporal stores, keeping the
// 192MB L2 free for the k/v operands that every workgroup re-reads).
// ---------------------------------------------------------------------------

typedef __attribute__((ext_vector_type(16))) __bf16 v16bf;
typedef __attribute__((ext_vector_type(8)))  __bf16 v8bf;
typedef __attribute__((ext_vector_type(8)))  float  v8f;

#define D_MODEL 512
#define N_HEAD  8
#define D_HEAD  64
#define BATCH   2
#define SEQ     4096
#define ROWS    (BATCH * SEQ)   // 8192

static __device__ __forceinline__ unsigned short f32_to_bf16_rne(float f) {
    unsigned int u = __float_as_uint(f);
    unsigned int r = u + 0x7FFFu + ((u >> 16) & 1u);
    return (unsigned short)(r >> 16);
}
static __device__ __forceinline__ __bf16 us_as_bf16(unsigned short u) {
    return __builtin_bit_cast(__bf16, u);
}
static __device__ __forceinline__ v16bf load16(const unsigned short* p) {
    return *reinterpret_cast<const v16bf*>(p);
}
static __device__ __forceinline__ v8bf load8(const unsigned short* p) {
    return *reinterpret_cast<const v8bf*>(p);
}
static __device__ __forceinline__ v16bf combine8(v8bf lo, v8bf hi) {
    v16bf a;
#pragma unroll
    for (int j = 0; j < 8; ++j) { a[j] = lo[j]; a[8 + j] = hi[j]; }
    return a;
}
static __device__ __forceinline__ v8f wmma_bf16(v16bf a, v16bf b, v8f c) {
    // (neg_a, A, neg_b, B, c_mod, C, reuse_a, reuse_b)
    return __builtin_amdgcn_wmma_f32_16x16x32_bf16(false, a, false, b,
                                                   (short)0, c, false, false);
}

// ---------------------------------------------------------------------------
// K0a: W[512x512] fp32 -> W^T bf16 (so GEMM B-tiles are row-contiguous)
// ---------------------------------------------------------------------------
__global__ void wt_transpose(const float* __restrict__ W,
                             unsigned short* __restrict__ WT) {
    int idx = blockIdx.x * blockDim.x + threadIdx.x;   // 0..262143
    int o = idx >> 9, i = idx & 511;
    WT[(size_t)o * D_MODEL + i] = f32_to_bf16_rne(W[(size_t)i * D_MODEL + o]);
}

// ---------------------------------------------------------------------------
// K0b: X fp32 -> bf16
// ---------------------------------------------------------------------------
__global__ void x_convert(const float* __restrict__ X,
                          unsigned short* __restrict__ Xbf, int n) {
    int idx = blockIdx.x * blockDim.x + threadIdx.x;
    if (idx < n) Xbf[idx] = f32_to_bf16_rne(X[idx]);
}

// ---------------------------------------------------------------------------
// K1: projection GEMM proj = Xbf @ W^T_bf + bias, scaled, stored head-split.
// One wave per 16x16 output tile; K-loop 512 in steps of 32 (16 WMMAs).
// A-fragment (16x32 bf16, ISA 7.12.2): lane L row M=L&15; lanes<16 hold
// K 0-7 & 16-23, lanes>=16 hold K 8-15 & 24-31.
// B-fragment (32x16): lane L column N=L&15; lanes<16 K 0-7 & 16-23, lanes>=16 K 8-15 & 24-31.
// layout 0: dst[b][h][s][d] bf16 (q is pre-scaled by 1/sqrt(64))
// layout 1: dst[b][h][d][s] bf16 (v transposed for the PV GEMM)
// ---------------------------------------------------------------------------
__global__ void __launch_bounds__(128)
gemm_proj(const unsigned short* __restrict__ Xbf,
          const unsigned short* __restrict__ WT,
          const float* __restrict__ bias,
          unsigned short* __restrict__ dst, int transposed, float scale) {
    const int lane = threadIdx.x & 31;
    const int wave = threadIdx.x >> 5;
    const int kh = lane >> 4;          // K-half select
    const int lm = lane & 15;
    const int rowBase = blockIdx.x * 16;
    const int colBase = (blockIdx.y * 4 + wave) * 16;

    const unsigned short* arow = Xbf + (size_t)(rowBase + lm) * D_MODEL;
    const unsigned short* brow = WT  + (size_t)(colBase + lm) * D_MODEL;

    v8f c = {};
#pragma unroll 4
    for (int kk = 0; kk < D_MODEL; kk += 32) {
        v16bf a = combine8(load8(arow + kk + kh * 8),
                           load8(arow + kk + 16 + kh * 8));
        v16bf b = combine8(load8(brow + kk + kh * 8),
                           load8(brow + kk + 16 + kh * 8));
        __builtin_prefetch(brow + kk + 64, 0, 1);   // global_prefetch_b8
        c = wmma_bf16(a, b, c);
    }
    const float bi = bias[colBase + lm];
#pragma unroll
    for (int i = 0; i < 8; ++i) {                    // C: M = i + 8*(lane>=16)
        float vo = (c[i] + bi) * scale;
        int r   = rowBase + i + 8 * kh;
        int col = colBase + lm;
        int bb = r >> 12, s = r & (SEQ - 1);
        int h = col >> 6, d = col & (D_HEAD - 1);
        size_t off = transposed
            ? ((size_t)(bb * N_HEAD + h) * D_HEAD + d) * SEQ + s
            : ((size_t)(bb * N_HEAD + h) * SEQ + s) * D_HEAD + d;
        dst[off] = f32_to_bf16_rne(vo);
    }
}

// ---------------------------------------------------------------------------
// K2: fused attention.  One workgroup = one (b,h) x 16 query rows.
// 256 threads = 8 waves (2 waves/SIMD even at 1 WG/WGP given 256KB LDS).
// Phase 1: 8 waves tile 16x4096 masked scores into LDS via WMMA (q@k^T).
// Phase 2: wave-per-row softmax in LDS; probs streamed to d_out with
//          nontemporal stores (don't pollute L2).
// Phase 3: PV GEMM; wave w computes d-slice (w&3) over key-half (w>>2);
//          halves reduced through an 8KB LDS staging region.
// LDS: 16*4096 fp32 = 256KB (dynamic), within the 320KB/WGP budget.
// ---------------------------------------------------------------------------
__global__ void __launch_bounds__(256)
attn_fused(const unsigned short* __restrict__ qh,
           const unsigned short* __restrict__ khd,
           const unsigned short* __restrict__ vT,
           const unsigned char* __restrict__ mask,
           float* __restrict__ attn_out,
           unsigned short* __restrict__ ctx) {
    extern __shared__ float sS[];                    // [16][SEQ]
    const int tid  = threadIdx.x;
    const int lane = tid & 31;
    const int wave = tid >> 5;                       // 0..7
    const int kh = lane >> 4;
    const int lm = lane & 15;
    const int qBase = blockIdx.x * 16;
    const int bh = blockIdx.y;                       // b*8 + h
    const int b  = bh >> 3;
    const int h  = bh & 7;

    // ---- phase 1: scores = (q * 1/sqrt(d)) @ k^T, masked, into LDS --------
    const unsigned short* qrow = qh + ((size_t)bh * SEQ + qBase + lm) * D_HEAD;
    v16bf a0 = combine8(load8(qrow + kh * 8),      load8(qrow + 16 + kh * 8));
    v16bf a1 = combine8(load8(qrow + 32 + kh * 8), load8(qrow + 48 + kh * 8));

    for (int kt = wave; kt < SEQ / 16; kt += 8) {
        const int keyBase = kt * 16;
        const unsigned short* krow =
            khd + ((size_t)bh * SEQ + keyBase + lm) * D_HEAD;
        v16bf b0 = combine8(load8(krow + kh * 8),      load8(krow + 16 + kh * 8));
        v16bf b1 = combine8(load8(krow + 32 + kh * 8), load8(krow + 48 + kh * 8));
        v8f c = {};
        c = wmma_bf16(a0, b0, c);
        c = wmma_bf16(a1, b1, c);
#pragma unroll
        for (int i = 0; i < 8; ++i) {
            int rr  = i + 8 * kh;
            int key = keyBase + lm;
            unsigned char m = __builtin_nontemporal_load(
                mask + ((size_t)b * SEQ + qBase + rr) * SEQ + key);
            sS[rr * SEQ + key] = m ? -1e9f : c[i];
        }
    }
    __syncthreads();

    // ---- phase 2: softmax per row (one wave per row, shfl reductions) -----
    for (int r = wave; r < 16; r += 8) {
        float mx = -3.4e38f;
        for (int cx = lane; cx < SEQ; cx += 32)
            mx = fmaxf(mx, sS[r * SEQ + cx]);
#pragma unroll
        for (int off = 16; off >= 1; off >>= 1)
            mx = fmaxf(mx, __shfl_xor(mx, off, 32));
        float sum = 0.f;
        for (int cx = lane; cx < SEQ; cx += 32)
            sum += __expf(sS[r * SEQ + cx] - mx);
#pragma unroll
        for (int off = 16; off >= 1; off >>= 1)
            sum += __shfl_xor(sum, off, 32);
        const float inv = 1.0f / sum;
        float* arow = attn_out + ((size_t)bh * SEQ + qBase + r) * SEQ;
        for (int cx = lane; cx < SEQ; cx += 32) {
            float p = __expf(sS[r * SEQ + cx] - mx) * inv;
            sS[r * SEQ + cx] = p;
            __builtin_nontemporal_store(p, arow + cx);  // only HBM prob write
        }
    }
    __syncthreads();

    // ---- phase 3: context = P @ V -----------------------------------------
    // wave w: d-slice (w&3), key half (w>>2); 64 WMMAs each.
    const int dslice = wave & 3;
    const int halfK  = (wave >> 2) * (SEQ / 2);
    const int dBase  = dslice * 16;
    const unsigned short* vrow = vT + ((size_t)bh * D_HEAD + dBase + lm) * SEQ;
    v8f c = {};
    for (int kk = halfK; kk < halfK + SEQ / 2; kk += 32) {
        const float* prow = sS + (size_t)lm * SEQ + kk + kh * 8;
        v16bf a;
#pragma unroll
        for (int j = 0; j < 8; ++j) {
            a[j]     = us_as_bf16(f32_to_bf16_rne(prow[j]));
            a[8 + j] = us_as_bf16(f32_to_bf16_rne(prow[16 + j]));
        }
        v16bf bm = combine8(load8(vrow + kk + kh * 8),
                            load8(vrow + kk + 16 + kh * 8));
        c = wmma_bf16(a, bm, c);
    }
    __syncthreads();     // all prob reads done; safe to reuse sS for partials

    // stage partial 16x16 C tiles: wave w -> sS[w*256 .. w*256+255]
#pragma unroll
    for (int i = 0; i < 8; ++i)
        sS[wave * 256 + (i + 8 * kh) * 16 + lm] = c[i];
    __syncthreads();

    // combine key-halves and store ctx bf16 [b][s][h*64+d]
    for (int e = tid; e < 1024; e += 256) {          // 16 rows x 64 cols
        int slice = e >> 8;                          // 0..3
        int rem   = e & 255;
        int row   = rem >> 4;
        int colIn = rem & 15;
        float val = sS[slice * 256 + rem] + sS[(slice + 4) * 256 + rem];
        int s   = qBase + row;
        int col = h * D_HEAD + slice * 16 + colIn;
        ctx[((size_t)b * SEQ + s) * D_MODEL + col] = f32_to_bf16_rne(val);
    }
}

// ---------------------------------------------------------------------------
// K3: output GEMM out = ctx_bf @ Wo^T_bf + bo  (fp32 result)
// ---------------------------------------------------------------------------
__global__ void __launch_bounds__(128)
gemm_out(const unsigned short* __restrict__ ctx,
         const unsigned short* __restrict__ WOT,
         const float* __restrict__ bo,
         float* __restrict__ out) {
    const int lane = threadIdx.x & 31;
    const int wave = threadIdx.x >> 5;
    const int kh = lane >> 4;
    const int lm = lane & 15;
    const int rowBase = blockIdx.x * 16;
    const int colBase = (blockIdx.y * 4 + wave) * 16;

    const unsigned short* arow = ctx + (size_t)(rowBase + lm) * D_MODEL;
    const unsigned short* brow = WOT + (size_t)(colBase + lm) * D_MODEL;

    v8f c = {};
#pragma unroll 4
    for (int kk = 0; kk < D_MODEL; kk += 32) {
        v16bf a = combine8(load8(arow + kk + kh * 8),
                           load8(arow + kk + 16 + kh * 8));
        v16bf b = combine8(load8(brow + kk + kh * 8),
                           load8(brow + kk + 16 + kh * 8));
        __builtin_prefetch(brow + kk + 64, 0, 1);
        c = wmma_bf16(a, b, c);
    }
    const float bi = bo[colBase + lm];
#pragma unroll
    for (int i = 0; i < 8; ++i) {
        int r   = rowBase + i + 8 * kh;
        int col = colBase + lm;
        out[(size_t)r * D_MODEL + col] = c[i] + bi;
    }
}

// ---------------------------------------------------------------------------
// Host launcher
// ---------------------------------------------------------------------------
extern "C" void kernel_launch(void* const* d_in, const int* in_sizes, int n_in,
                              void* d_out, int out_size, void* d_ws,
                              size_t ws_size, hipStream_t stream) {
    (void)in_sizes; (void)n_in; (void)out_size; (void)ws_size;

    const float* Q  = (const float*)d_in[0];
    const float* K  = (const float*)d_in[1];
    const float* V  = (const float*)d_in[2];
    const unsigned char* mask = (const unsigned char*)d_in[3];  // jnp bool_
    const float* Wq = (const float*)d_in[4];
    const float* bq = (const float*)d_in[5];
    const float* Wk = (const float*)d_in[6];
    const float* bk = (const float*)d_in[7];
    const float* Wv = (const float*)d_in[8];
    const float* bv = (const float*)d_in[9];
    const float* Wo = (const float*)d_in[10];
    const float* bo = (const float*)d_in[11];

    float* out  = (float*)d_out;                               // [B,S,512]
    float* attn = out + (size_t)BATCH * SEQ * D_MODEL;         // [B,H,S,S]

    // workspace carve-up (bf16 elements); total ~58MB
    unsigned short* w = (unsigned short*)d_ws;
    const size_t WSZ = (size_t)D_MODEL * D_MODEL;              // 262144
    const size_t XSZ = (size_t)ROWS * D_MODEL;                 // 4194304
    const size_t HSZ = (size_t)BATCH * N_HEAD * SEQ * D_HEAD;  // 4194304
    unsigned short* WQT = w;
    unsigned short* WKT = WQT + WSZ;
    unsigned short* WVT = WKT + WSZ;
    unsigned short* WOT = WVT + WSZ;
    unsigned short* QBF = WOT + WSZ;
    unsigned short* KBF = QBF + XSZ;
    unsigned short* VBF = KBF + XSZ;
    unsigned short* qhp = VBF + XSZ;   // [b,h,s,d] bf16, pre-scaled
    unsigned short* khp = qhp + HSZ;   // [b,h,s,d] bf16
    unsigned short* vTp = khp + HSZ;   // [b,h,d,s] bf16
    unsigned short* ctxp = vTp + HSZ;  // [b,s,512] bf16

    // K0: weight transpose+convert and input convert
    wt_transpose<<<dim3(1024), 256, 0, stream>>>(Wq, WQT);
    wt_transpose<<<dim3(1024), 256, 0, stream>>>(Wk, WKT);
    wt_transpose<<<dim3(1024), 256, 0, stream>>>(Wv, WVT);
    wt_transpose<<<dim3(1024), 256, 0, stream>>>(Wo, WOT);
    x_convert<<<dim3((int)(XSZ / 256)), 256, 0, stream>>>(Q, QBF, (int)XSZ);
    x_convert<<<dim3((int)(XSZ / 256)), 256, 0, stream>>>(K, KBF, (int)XSZ);
    x_convert<<<dim3((int)(XSZ / 256)), 256, 0, stream>>>(V, VBF, (int)XSZ);

    // K1: projections (q scaled by 1/sqrt(D_HEAD), v stored transposed)
    gemm_proj<<<dim3(ROWS / 16, 8), 128, 0, stream>>>(QBF, WQT, bq, qhp, 0, 0.125f);
    gemm_proj<<<dim3(ROWS / 16, 8), 128, 0, stream>>>(KBF, WKT, bk, khp, 0, 1.0f);
    gemm_proj<<<dim3(ROWS / 16, 8), 128, 0, stream>>>(VBF, WVT, bv, vTp, 1, 1.0f);

    // K2: fused scores+softmax+context; 256KB dynamic LDS per workgroup
    attn_fused<<<dim3(SEQ / 16, BATCH * N_HEAD), 256,
                 16 * SEQ * sizeof(float), stream>>>(qhp, khp, vTp, mask,
                                                     attn, ctxp);

    // K3: output projection (fp32 result)
    gemm_out<<<dim3(ROWS / 16, 8), 128, 0, stream>>>(ctxp, WOT, bo, out);
}
